// SurfaceDeform_42228118454927
// MI455X (gfx1250) — matrix operand order repaired and checked by reading, not compiled, not run on hardware
//
#include <hip/hip_runtime.h>
#include <hip/hip_bf16.h>

// ---------------- problem constants (from reference) ----------------
#define B_   2
#define V_   100000
#define NTOT 200000            // B*V
#define E_   1200000
#define CIN0 155               // 3 + 32 + (8+16+32+64)
#define KPAD0 160              // CIN0 padded to multiple of 32 for WMMA K
#define COUT 64

typedef __attribute__((ext_vector_type(16))) _Float16 v16h;
typedef __attribute__((ext_vector_type(8)))  _Float16 v8h;
typedef __attribute__((ext_vector_type(8)))  float    v8f;

union U16h { v16h v; v8h h[2]; };

// ---------------- utility ----------------
__global__ void k_zero_f32(float* __restrict__ p, size_t n) {
    size_t i = (size_t)blockIdx.x * blockDim.x + threadIdx.x;
    if (i < n) p[i] = 0.0f;
}

__global__ void k_degree(const int* __restrict__ edges, unsigned* __restrict__ deg) {
    int e = blockIdx.x * blockDim.x + threadIdx.x;
    if (e >= E_) return;
    atomicAdd(&deg[edges[2 * e + 0]], 1u);
    atomicAdd(&deg[edges[2 * e + 1]], 1u);
}

__global__ void k_invdeg(const unsigned* __restrict__ deg, float* __restrict__ invd) {
    int n = blockIdx.x * blockDim.x + threadIdx.x;
    if (n >= NTOT) return;
    invd[n] = 1.0f / fmaxf((float)deg[n], 1.0f);
}

// fp32 -> f16 weight conversion with zero K-padding, written DIRECTLY in the
// WMMA B-fragment packed order so the GEMM can async-copy it linearly to LDS.
// Packed order per (kt,ntile) tile of 512 halves:
//   lane = n2 + (k2 & 16)  (lanes 0-15 hold K 0..15, lanes 16-31 hold K 16..31)
//   slot = lane*16 + (k2 & 15)
__global__ void k_cvt_w(const float* __restrict__ src, _Float16* __restrict__ dst,
                        int Ksrc, int Kpad) {
    int i = blockIdx.x * blockDim.x + threadIdx.x;
    if (i >= Kpad * COUT) return;
    int k = i >> 6, n = i & 63;
    int kt = k >> 5, k2 = k & 31, ntile = n >> 4, n2 = n & 15;
    int lane = n2 + (k2 & 16);
    int d = ((kt << 2) + ntile) * 512 + lane * 16 + (k2 & 15);
    dst[d] = (k < Ksrc) ? (_Float16)src[i] : (_Float16)0.0f;
}

// ---------------- trilinear sample + feature concat (wave per node) ----------------
__device__ __forceinline__ float pixf(float p, int size) {
    float t = (p + 1.0f) * 0.5f * (float)(size - 1);
    return fminf(fmaxf(t, 0.0f), (float)(size - 1));
}

__global__ __launch_bounds__(128)
void k_sample(const float* __restrict__ Vt, const float* __restrict__ gin,
              const float* __restrict__ s0, const float* __restrict__ s1,
              const float* __restrict__ s2, const float* __restrict__ s3,
              _Float16* __restrict__ xh) {
    int n = blockIdx.x * 4 + (threadIdx.x >> 5);
    if (n >= NTOT) return;
    int lane = threadIdx.x & 31;
    int b = n / V_;

    float p0 = Vt[(size_t)n * 3 + 0];
    float p1 = Vt[(size_t)n * 3 + 1];
    float p2 = Vt[(size_t)n * 3 + 2];

    const int Sv[4] = {128, 64, 32, 16};
    int   so[4][8];
    float w8[4][8];
    #pragma unroll
    for (int q = 0; q < 4; ++q) {
        int S = Sv[q];
        float x = pixf(p0, S), y = pixf(p1, S), z = pixf(p2, S);
        float fx = floorf(x), fy = floorf(y), fz = floorf(z);
        int ix0 = (int)fx, iy0 = (int)fy, iz0 = (int)fz;
        int ix1 = min(ix0 + 1, S - 1), iy1 = min(iy0 + 1, S - 1), iz1 = min(iz0 + 1, S - 1);
        float wx = x - fx, wy = y - fy, wz = z - fz;
        so[q][0] = (iz0 * S + iy0) * S + ix0;
        so[q][1] = (iz0 * S + iy0) * S + ix1;
        so[q][2] = (iz0 * S + iy1) * S + ix0;
        so[q][3] = (iz0 * S + iy1) * S + ix1;
        so[q][4] = (iz1 * S + iy0) * S + ix0;
        so[q][5] = (iz1 * S + iy0) * S + ix1;
        so[q][6] = (iz1 * S + iy1) * S + ix0;
        so[q][7] = (iz1 * S + iy1) * S + ix1;
        w8[q][0] = (1 - wz) * (1 - wy) * (1 - wx);
        w8[q][1] = (1 - wz) * (1 - wy) * wx;
        w8[q][2] = (1 - wz) * wy * (1 - wx);
        w8[q][3] = (1 - wz) * wy * wx;
        w8[q][4] = wz * (1 - wy) * (1 - wx);
        w8[q][5] = wz * (1 - wy) * wx;
        w8[q][6] = wz * wy * (1 - wx);
        w8[q][7] = wz * wy * wx;
    }

    for (int col = lane; col < KPAD0; col += 32) {
        float val;
        if (col < 3) {
            val = (col == 0) ? p0 : (col == 1 ? p1 : p2);
        } else if (col < 35) {
            val = gin[(size_t)n * 32 + (col - 3)];
        } else if (col < CIN0) {
            int q, ch, C;
            const float* vol;
            if (col < 43)       { q = 0; ch = col - 35; C = 8;  vol = s0; }
            else if (col < 59)  { q = 1; ch = col - 43; C = 16; vol = s1; }
            else if (col < 91)  { q = 2; ch = col - 59; C = 32; vol = s2; }
            else                { q = 3; ch = col - 91; C = 64; vol = s3; }
            int S = Sv[q];
            long vb = ((long)(b * C + ch)) * S * S * S;
            float acc = 0.0f;
            #pragma unroll
            for (int c8 = 0; c8 < 8; ++c8) acc += vol[vb + so[q][c8]] * w8[q][c8];
            val = acc;
        } else {
            val = 0.0f; // K padding
        }
        xh[(size_t)n * KPAD0 + col] = (_Float16)val;
    }
}

// ---------------- dual WMMA GEMM: Y0 = X@W0, Y1 = X@W1 (f16 in, f32 out) ----------------
// X: NTOT x K (f16, row stride = ldx). Wg: BOTH weight matrices, pre-packed in
// B-fragment order, contiguous (W0 then W1), 2*K*64 halves. Weights are staged
// into LDS with CDNA5 async global->LDS b128 copies (ASYNCcnt path).
// Block = 128 threads = 4 waves; each wave owns a 16-row tile, 4 N-tiles of 16.
__global__ __launch_bounds__(128)
void k_gemm_dual(const _Float16* __restrict__ X, int ldx, int K,
                 const _Float16* __restrict__ Wg,
                 float* __restrict__ Y0, float* __restrict__ Y1) {
    __shared__ _Float16 sW[2 * KPAD0 * COUT]; // 40 KB max, already fragment-packed

    const int tid = threadIdx.x;
    const int KN  = K * COUT;

    // Async stage both weight matrices (linear copy: global packed -> LDS packed).
    {
        const int totBytes = 2 * KN * 2;                 // 16KB (K=64) or 40KB (K=160)
        unsigned sbase = (unsigned)(uintptr_t)(&sW[0]);  // LDS aperture: low 32 bits = LDS addr
        const char* gsrc = (const char*)Wg;
        for (int i = tid * 16; i < totBytes; i += 128 * 16) {
            unsigned ldst = sbase + (unsigned)i;
            const char* gp = gsrc + i;
            asm volatile("global_load_async_to_lds_b128 %0, %1, off"
                         :: "v"(ldst), "v"(gp) : "memory");
        }
        asm volatile("s_wait_asynccnt 0x0" ::: "memory");
    }
    __syncthreads();

    const int lane  = tid & 31;
    const int wave  = tid >> 5;
    const int m0    = (blockIdx.x << 6) + (wave << 4);
    const int half8 = (lane >> 4) << 4; // lanes 16-31 take K 16..31
    const _Float16* xrow = X + (size_t)(m0 + (lane & 15)) * ldx;

    v8f acc0[4] = {{0}}; v8f acc1[4] = {{0}};

    const int ktiles = K >> 5;
    for (int kt = 0; kt < ktiles; ++kt) {
        const int k0 = kt << 5;
        U16h a;
        // A fragment, contiguous k order chosen here: lane<16 -> K 0..15; lane>=16 -> K 16..31 (matches packed B)
        a.h[0] = *(const v8h*)(xrow + k0 + half8);
        a.h[1] = *(const v8h*)(xrow + k0 + 8 + half8);
        if (kt + 1 < ktiles)
            __builtin_prefetch(xrow + k0 + 32, 0, 1); // global_prefetch_b8

        const _Float16* sp = sW + (kt << 2) * 512 + lane * 16;
        #pragma unroll
        for (int nt = 0; nt < 4; ++nt) {
            U16h b0, b1;
            b0.h[0] = *(const v8h*)(sp + nt * 512);
            b0.h[1] = *(const v8h*)(sp + nt * 512 + 8);
            b1.h[0] = *(const v8h*)(sp + KN + nt * 512);
            b1.h[1] = *(const v8h*)(sp + KN + nt * 512 + 8);
            acc0[nt] = __builtin_amdgcn_wmma_f32_16x16x32_f16(
                false, a.v, false, b0.v, (short)0, acc0[nt], false, false);
            acc1[nt] = __builtin_amdgcn_wmma_f32_16x16x32_f16(
                false, a.v, false, b1.v, (short)0, acc1[nt], false, false);
        }
    }

    // C/D layout: VGPR r, lanes 0-15 -> (M=r, N=lane); lanes 16-31 -> (M=r+8, N=lane-16)
    const int nbase = lane & 15;
    const int rofs  = (lane >> 4) << 3;
    #pragma unroll
    for (int nt = 0; nt < 4; ++nt) {
        #pragma unroll
        for (int r = 0; r < 8; ++r) {
            size_t o = (size_t)(m0 + r + rofs) * COUT + nt * 16 + nbase;
            Y0[o] = acc0[nt][r];
            Y1[o] = acc1[nt][r];
        }
    }
}

// ---------------- edge scatter (wave per edge, 64 channels, L2-resident) ----------------
__global__ __launch_bounds__(128)
void k_scatter64(const int* __restrict__ edges, const float* __restrict__ m,
                 float* __restrict__ agg) {
    int gid = blockIdx.x * blockDim.x + threadIdx.x;
    int e = gid >> 5;
    if (e >= E_) return;
    int lane = gid & 31;
    int e0 = edges[2 * e + 0];
    int e1 = edges[2 * e + 1];
    const float* m0p = m + (size_t)e0 * COUT;
    const float* m1p = m + (size_t)e1 * COUT;
    float* a0p = agg + (size_t)e0 * COUT;
    float* a1p = agg + (size_t)e1 * COUT;
    atomicAdd(a0p + lane,      m1p[lane]);
    atomicAdd(a0p + lane + 32, m1p[lane + 32]);
    atomicAdd(a1p + lane,      m0p[lane]);
    atomicAdd(a1p + lane + 32, m0p[lane + 32]);
}

// ---------------- gconv epilogue: bias + agg*invdeg, layernorm, relu, residual ----------------
__global__ __launch_bounds__(128)
void k_epilogue(const float* __restrict__ Y0, const float* __restrict__ agg,
                const float* __restrict__ invd, const float* __restrict__ bias,
                const float* __restrict__ gamma, const float* __restrict__ beta,
                float* __restrict__ res, _Float16* __restrict__ xh,
                int storeRes, int addRes) {
    int n = blockIdx.x * 4 + (threadIdx.x >> 5);
    if (n >= NTOT) return;
    int lane = threadIdx.x & 31;
    int c0 = lane, c1 = lane + 32;
    size_t o = (size_t)n * COUT;
    float id = invd[n];

    float a0 = Y0[o + c0] + bias[c0] + agg[o + c0] * id;
    float a1 = Y0[o + c1] + bias[c1] + agg[o + c1] * id;

    float s = a0 + a1;
    #pragma unroll
    for (int off = 16; off > 0; off >>= 1) s += __shfl_xor(s, off, 32);
    float mean = s * (1.0f / 64.0f);
    float q = a0 * a0 + a1 * a1;
    #pragma unroll
    for (int off = 16; off > 0; off >>= 1) q += __shfl_xor(q, off, 32);
    float var = q * (1.0f / 64.0f) - mean * mean;
    float rs = rsqrtf(var + 1e-5f);

    float h0 = fmaxf((a0 - mean) * rs * gamma[c0] + beta[c0], 0.0f);
    float h1 = fmaxf((a1 - mean) * rs * gamma[c1] + beta[c1], 0.0f);
    if (storeRes) { res[o + c0] = h0; res[o + c1] = h1; }
    if (addRes)   { h0 += res[o + c0]; h1 += res[o + c1]; }
    xh[o + c0] = (_Float16)h0;   // next layer reads f16, stride 64
    xh[o + c1] = (_Float16)h1;
}

// ---------------- final f2v gconv (64 -> 3) ----------------
__global__ void k_f2v_m(const _Float16* __restrict__ xh, const float* __restrict__ w1,
                        float* __restrict__ m2) {
    int i = blockIdx.x * blockDim.x + threadIdx.x;
    if (i >= NTOT * 3) return;
    int n = i / 3, d = i - n * 3;
    const _Float16* xr = xh + (size_t)n * COUT;
    float acc = 0.0f;
    for (int k = 0; k < COUT; ++k) acc += (float)xr[k] * w1[k * 3 + d];
    m2[i] = acc;
}

__global__ void k_scatter3(const int* __restrict__ edges, const float* __restrict__ m2,
                           float* __restrict__ aggV) {
    int e = blockIdx.x * blockDim.x + threadIdx.x;
    if (e >= E_) return;
    int e0 = edges[2 * e + 0];
    int e1 = edges[2 * e + 1];
    #pragma unroll
    for (int d = 0; d < 3; ++d) {
        atomicAdd(&aggV[(size_t)e0 * 3 + d], m2[(size_t)e1 * 3 + d]);
        atomicAdd(&aggV[(size_t)e1 * 3 + d], m2[(size_t)e0 * 3 + d]);
    }
}

__global__ void k_f2v_final(const _Float16* __restrict__ xh, const float* __restrict__ w0,
                            const float* __restrict__ bias, const float* __restrict__ aggV,
                            const float* __restrict__ invd, float* __restrict__ out) {
    int i = blockIdx.x * blockDim.x + threadIdx.x;
    if (i >= NTOT * 3) return;
    int n = i / 3, d = i - n * 3;
    const _Float16* xr = xh + (size_t)n * COUT;
    float acc = 0.0f;
    for (int k = 0; k < COUT; ++k) acc += (float)xr[k] * w0[k * 3 + d];
    out[i] = acc + bias[d] + aggV[i] * invd[n];
}

// ---------------- host launch ----------------
extern "C" void kernel_launch(void* const* d_in, const int* in_sizes, int n_in,
                              void* d_out, int out_size, void* d_ws, size_t ws_size,
                              hipStream_t stream) {
    const float* Vt  = (const float*)d_in[0];
    const float* gin = (const float*)d_in[1];
    const float* s0  = (const float*)d_in[2];
    const float* s1  = (const float*)d_in[3];
    const float* s2  = (const float*)d_in[4];
    const float* s3  = (const float*)d_in[5];
    const int*   edges = (const int*)d_in[6];

    // ---- workspace carve-out ----
    char* ws = (char*)d_ws;
    size_t off = 0;
    auto carve = [&](size_t bytes) -> void* {
        void* p = ws + off;
        off = (off + bytes + 255) & ~(size_t)255;
        return p;
    };
    _Float16* xh   = (_Float16*)carve((size_t)NTOT * KPAD0 * 2);
    float*    Y0   = (float*)   carve((size_t)NTOT * COUT * 4);
    float*    Y1   = (float*)   carve((size_t)NTOT * COUT * 4);
    float*    agg  = (float*)   carve((size_t)NTOT * COUT * 4);
    float*    res  = (float*)   carve((size_t)NTOT * COUT * 4);
    float*    invd = (float*)   carve((size_t)NTOT * 4);
    unsigned* deg  = (unsigned*)carve((size_t)NTOT * 4);
    float*    m2   = (float*)   carve((size_t)NTOT * 3 * 4);
    float*    aggV = (float*)   carve((size_t)NTOT * 3 * 4);
    _Float16* wh   = (_Float16*)carve((size_t)(2 * KPAD0 * COUT + 8 * 2 * COUT * COUT) * 2);

    // ---- resolve param leaves (jax tree order: dict keys sorted) ----
    // per layer: beta(64), gamma(64), gc.b(64), gc.w0(cin*64), gc.w1(cin*64); then f2v: b, w0, w1
    const float* leaf[48];
    int cins[9];
    for (int l = 0; l < 9; ++l) cins[l] = (l == 0) ? CIN0 : COUT;
    size_t lsz[48]; int li = 0;
    for (int l = 0; l < 9; ++l) {
        lsz[li++] = 64; lsz[li++] = 64; lsz[li++] = 64;
        lsz[li++] = (size_t)cins[l] * 64; lsz[li++] = (size_t)cins[l] * 64;
    }
    lsz[45] = 3; lsz[46] = 192; lsz[47] = 192;
    if (n_in >= 55) {
        for (int i = 0; i < 48; ++i) leaf[i] = (const float*)d_in[7 + i];
    } else {
        const float* base = (const float*)d_in[7];
        size_t o = 0;
        for (int i = 0; i < 48; ++i) { leaf[i] = base + o; o += lsz[i]; }
    }

    // ---- degrees ----
    k_zero_f32<<<(NTOT + 255) / 256, 256, 0, stream>>>((float*)deg, NTOT);
    k_degree<<<(E_ + 255) / 256, 256, 0, stream>>>(edges, deg);
    k_invdeg<<<(NTOT + 255) / 256, 256, 0, stream>>>(deg, invd);

    // ---- build x0 (trilinear sample + concat) as f16, K padded to 160 ----
    k_sample<<<(NTOT + 3) / 4, 128, 0, stream>>>(Vt, gin, s0, s1, s2, s3, xh);

    // ---- convert weights to f16, pre-packed in WMMA B-fragment order ----
    // W0 and W1 of each layer are contiguous so the GEMM async-copies one block.
    _Float16* wpacked[9];
    size_t wo = 0;
    for (int l = 0; l < 9; ++l) {
        int Kp = (l == 0) ? KPAD0 : COUT;
        wpacked[l] = wh + wo;
        for (int m = 0; m < 2; ++m) {
            k_cvt_w<<<(Kp * COUT + 255) / 256, 256, 0, stream>>>(
                leaf[l * 5 + 3 + m], wh + wo, cins[l], Kp);
            wo += (size_t)Kp * COUT;
        }
    }

    // ---- 9 graph-conv layers ----
    for (int l = 0; l < 9; ++l) {
        int Kp = (l == 0) ? KPAD0 : COUT;
        k_gemm_dual<<<NTOT / 64, 128, 0, stream>>>(xh, Kp, Kp, wpacked[l], Y0, Y1);
        k_zero_f32<<<((NTOT * COUT) + 255) / 256, 256, 0, stream>>>(agg, (size_t)NTOT * COUT);
        k_scatter64<<<(E_ * 32 + 127) / 128, 128, 0, stream>>>(edges, Y1, agg);
        int il = l % 3;
        k_epilogue<<<(NTOT + 3) / 4, 128, 0, stream>>>(
            Y0, agg, invd,
            leaf[l * 5 + 2],  // gc.b
            leaf[l * 5 + 1],  // gamma
            leaf[l * 5 + 0],  // beta
            res, xh, il == 0, il == 2);
    }

    // ---- final f2v gconv (64 -> 3) ----
    k_f2v_m<<<(NTOT * 3 + 255) / 256, 256, 0, stream>>>(xh, leaf[47], m2);
    k_zero_f32<<<(NTOT * 3 + 255) / 256, 256, 0, stream>>>(aggV, (size_t)NTOT * 3);
    k_scatter3<<<(E_ + 255) / 256, 256, 0, stream>>>(edges, m2, aggV);
    k_f2v_final<<<(NTOT * 3 + 255) / 256, 256, 0, stream>>>(
        xh, leaf[46], leaf[45], aggV, invd, (float*)d_out);
}
